// WaveletViTBlockND_75806172774849
// MI455X (gfx1250) — matrix operand order repaired and checked by reading, not compiled, not run on hardware
//
#include <hip/hip_runtime.h>

// ---------------------------------------------------------------------------
// CDNA5 (gfx1250) implementation of the WaveletViT block.
// All GEMM-shaped math runs through v_wmma_f32_16x16x32_f16 (wave32).
// ---------------------------------------------------------------------------

typedef __attribute__((ext_vector_type(16))) _Float16 v16h;
typedef __attribute__((ext_vector_type(8)))  _Float16 v8h;
typedef __attribute__((ext_vector_type(8)))  float    v8f;

union V16H { v16h v; v8h h[2]; _Float16 e[16]; };
union V8F  { v8f  v; float f[8]; };
union PK2  { _Float16 h[2]; unsigned u; };

#define CCH 96      // channels
#define HD  128     // heads*kd
#define NHEAD 4
#define KD  32

// --- WMMA wrapper -----------------------------------------------------------
__device__ __forceinline__ v8f wmma32(v16h a, v16h b, v8f c) {
  return __builtin_amdgcn_wmma_f32_16x16x32_f16(
      /*neg_a=*/false, a, /*neg_b=*/false, b, /*c_mod=*/(short)0, c,
      /*reuse_a=*/false, /*reuse_b=*/false);
}

// --- A fragment (16x32 f16, M x K) from a row-major matrix -------------------
// Layout: lane 0-15 row M=lane, K per elem e: 16*(lane>=16) + e (contiguous).
// => two contiguous 8-f16 chunks at k0 + 16*(lane>=16) and +8.
__device__ __forceinline__ v16h afrag_rm(const _Float16* M, int ld, int row0,
                                         int k0, int lane) {
  const _Float16* p = M + (row0 + (lane & 15)) * ld + k0 + ((lane >> 4) << 4);
  V16H u;
  u.h[0] = *(const v8h*)(p);
  u.h[1] = *(const v8h*)(p + 8);
  return u.v;
}

// --- B fragment (32x16 f16, K x N) where source S is row-major [N][ld] -------
// (i.e. B = S^T). Lane holds column N = n0+(lane&15); K = 16*(lane>=16)+e,
// contiguous -> two contiguous 8-f16 chunks.
__device__ __forceinline__ v16h bfrag_nt(const _Float16* S, int ld, int n0,
                                         int k0, int lane) {
  const _Float16* p = S + (n0 + (lane & 15)) * ld + k0 + ((lane >> 4) << 4);
  V16H u;
  u.h[0] = *(const v8h*)(p);
  u.h[1] = *(const v8h*)(p + 8);
  return u.v;
}

// --- B fragment from pre-packed weights (fragment-ordered, 16 f16 per lane) --
__device__ __forceinline__ v16h bfrag_packed(const _Float16* P, int frag,
                                             int lane) {
  const _Float16* p = P + (((long)frag << 5) + lane) * 16;
  V16H u;
  u.h[0] = *(const v8h*)(p);
  u.h[1] = *(const v8h*)(p + 8);
  return u.v;
}

// token -> element index of channel 0 in the [B,H,W,C] tensor.
// strided==0: seq-major contiguous ((b*H+h) rows attending along W, or global)
// strided==1: axis-1 pass, seq=(b*W+w), token = h.
__device__ __forceinline__ int tok_base(int strided, int Hd, int Wd, int T,
                                        int seq, int t) {
  if (!strided) return (seq * T + t) * CCH;
  int b = seq / Wd;
  int w = seq - b * Wd;
  return ((b * Hd + t) * Wd + w) * CCH;
}

// ---------------------------------------------------------------------------
// Pack fp32 weight matrix [Kdim][Ndim] into f16 B-fragment order.
// grid = (Kdim/32)*(Ndim/16) blocks of 32 lanes.
// ---------------------------------------------------------------------------
__global__ void pack_w_kernel(const float* __restrict__ W,
                              _Float16* __restrict__ dst, int Kdim, int Ndim) {
  int frag = blockIdx.x;
  int lane = threadIdx.x;
  int nTiles = Ndim >> 4;
  int kt = frag / nTiles;
  int nt = frag - kt * nTiles;
  int n = nt * 16 + (lane & 15);
  int kb = kt * 32 + ((lane >> 4) << 4);
#pragma unroll
  for (int e = 0; e < 16; e++) {
    dst[((long)frag * 32 + lane) * 16 + e] = (_Float16)W[(kb + e) * Ndim + n];
  }
}

// ---------------------------------------------------------------------------
// LayerNorm + QKV projection for one 16-token tile. 96 threads = 3 waves:
// wave0 -> Q, wave1 -> K, wave2 -> V (stored transposed Vt[d][t]).
// ---------------------------------------------------------------------------
__global__ void ln_qkv_kernel(const float* __restrict__ x,
                              const float* __restrict__ gw,
                              const float* __restrict__ bw,
                              const _Float16* __restrict__ WqP,
                              const float* __restrict__ bq,
                              const _Float16* __restrict__ WkP,
                              const float* __restrict__ bk,
                              const _Float16* __restrict__ WvP,
                              const float* __restrict__ bv,
                              _Float16* __restrict__ Qm,
                              _Float16* __restrict__ Km,
                              _Float16* __restrict__ Vt,
                              int T, int Tpad, int Hd, int Wd, int strided) {
  __shared__ __align__(16) _Float16 Xh[16 * CCH];
  __shared__ float red[16][12];
  __shared__ float stats[16][2];

  int strips = T >> 4;
  int seq = blockIdx.x / strips;
  int t0 = (blockIdx.x - seq * strips) << 4;
  int tid = threadIdx.x;
  int tok = tid & 15;
  int part = tid >> 4;  // 0..5, 16 channels each

  int base = tok_base(strided, Hd, Wd, T, seq, t0 + tok);
  float s = 0.f, s2 = 0.f;
#pragma unroll
  for (int c = 0; c < 16; c++) {
    float v = x[base + part * 16 + c];
    s += v; s2 += v * v;
  }
  red[tok][part * 2 + 0] = s;
  red[tok][part * 2 + 1] = s2;
  __syncthreads();
  if (tid < 16) {
    float ss = 0.f, qq = 0.f;
#pragma unroll
    for (int p = 0; p < 6; p++) { ss += red[tid][2 * p]; qq += red[tid][2 * p + 1]; }
    float mean = ss * (1.f / 96.f);
    float var = qq * (1.f / 96.f) - mean * mean;
    stats[tid][0] = mean;
    stats[tid][1] = rsqrtf(var + 1e-5f);
  }
  __syncthreads();
  float mean = stats[tok][0], inv = stats[tok][1];
#pragma unroll
  for (int c = 0; c < 16; c++) {
    int ch = part * 16 + c;
    float v = (x[base + ch] - mean) * inv * gw[ch] + bw[ch];
    Xh[tok * CCH + ch] = (_Float16)v;
  }
  __syncthreads();

  int wv = tid >> 5;
  int lane = tid & 31;
  const _Float16* WP = (wv == 0) ? WqP : (wv == 1) ? WkP : WvP;
  const float* bias = (wv == 0) ? bq : (wv == 1) ? bk : bv;
  int m0 = (lane >> 4) << 3;

#pragma unroll
  for (int nt = 0; nt < 8; nt++) {
    V8F acc = {};
#pragma unroll
    for (int kt = 0; kt < 3; kt++) {
      v16h a = afrag_rm(Xh, CCH, 0, kt * 32, lane);
      v16h b = bfrag_packed(WP, kt * 8 + nt, lane);
      acc.v = wmma32(a, b, acc.v);
    }
    float bb = bias[nt * 16 + (lane & 15)];
    if (wv == 2) {  // V, transposed: Vt[seq][d][t]
      long vb = (long)seq * HD * Tpad + (nt * 16 + (lane & 15)) * Tpad + t0 + m0;
#pragma unroll
      for (int g = 0; g < 8; g++) Vt[vb + g] = (_Float16)(acc.f[g] + bb);
    } else {
      _Float16* dst = (wv == 0) ? Qm : Km;
      long rb = ((long)seq * Tpad + t0 + m0) * HD + nt * 16 + (lane & 15);
#pragma unroll
      for (int g = 0; g < 8; g++) dst[rb + (long)g * HD] = (_Float16)(acc.f[g] + bb);
    }
  }
}

// ---------------------------------------------------------------------------
// Flash-style attention. One wave per (seq, head, 16-query strip).
// Keys processed in chunks of 96 (Tpad is a multiple of 96; pad keys are
// zeroed in K/Vt and masked out of the softmax).
// Softmax is fully lane-parallel: each lane owns a 48-column half-row;
// partials are combined across the lane pair with __shfl_xor(.,16).
// ---------------------------------------------------------------------------
__global__ void attn_kernel(const _Float16* __restrict__ Qm,
                            const _Float16* __restrict__ Km,
                            const _Float16* __restrict__ Vt,
                            _Float16* __restrict__ Ym, int T, int Tpad) {
  __shared__ __align__(16) float sS[16 * 96];
  __shared__ __align__(16) _Float16 sP[16 * 96];
  __shared__ float rowScale[16];
  __shared__ float rowSum[16];

  int lane = threadIdx.x;
  int strips = T >> 4;
  int strip = blockIdx.x % strips;
  int head = (blockIdx.x / strips) & 3;
  int seq = blockIdx.x / (strips * NHEAD);
  int q0 = strip << 4;

  const _Float16* Qs = Qm + (long)seq * Tpad * HD + head * KD;
  const _Float16* Ks = Km + (long)seq * Tpad * HD + head * KD;
  const _Float16* Vs = Vt + (long)seq * HD * Tpad + (long)head * KD * Tpad;

  V8F acc0 = {}, acc1 = {};
  float rM = -3.0e38f;   // running max (identical within a lane pair)
  float rSp = 0.f;       // running sum, per-lane partial (half a row)
  int m0 = (lane >> 4) << 3;
  int row = lane & 15;
  int half = lane >> 4;
  float* srow = &sS[row * 96 + half * 48];
  _Float16* prow = &sP[row * 96 + half * 48];

  v16h aq = afrag_rm(Qs, HD, q0, 0, lane);  // Q fragment, constant per strip

  for (int s0 = 0; s0 < Tpad; s0 += 96) {
    // --- scores: 16 x 96 via 6 WMMAs -------------------------------------
#pragma unroll
    for (int j = 0; j < 6; j++) {
      V8F c = {};
      v16h b = bfrag_nt(Ks, HD, s0 + j * 16, 0, lane);
      c.v = wmma32(aq, b, c.v);
      int n = j * 16 + (lane & 15);
#pragma unroll
      for (int g = 0; g < 8; g++)
        sS[(m0 + g) * 96 + n] = c.f[g] * 0.17677669529663687f;  // 1/sqrt(32)
    }
    __syncthreads();

    // --- online softmax, all 32 lanes (48 columns each) ------------------
    int L = T - s0; if (L > 96) L = 96;            // valid keys this chunk
    int lv = L - half * 48;                        // valid in this half-row
    lv = lv < 0 ? 0 : (lv > 48 ? 48 : lv);
    float cmax = -3.0e38f;
    for (int j = 0; j < lv; j++) cmax = fmaxf(cmax, srow[j]);
    cmax = fmaxf(cmax, __shfl_xor(cmax, 16));      // combine lane pair
    float mnew = fmaxf(rM, cmax);
    float scale = __expf(rM - mnew);
    float ssum = 0.f;
    for (int j = 0; j < lv; j += 2) {
      float p0 = __expf(srow[j] - mnew);
      float p1 = __expf(srow[j + 1] - mnew);
      ssum += p0 + p1;
      PK2 pk; pk.h[0] = (_Float16)p0; pk.h[1] = (_Float16)p1;
      *(unsigned*)(prow + j) = pk.u;               // packed f16 pair store
    }
    for (int j = lv; j < 48; j += 2) *(unsigned*)(prow + j) = 0u;
    rSp = rSp * scale + ssum;
    rM = mnew;
    if (lane < 16) rowScale[lane] = scale;
    __syncthreads();

    // --- rescale accumulators, then P @ V --------------------------------
#pragma unroll
    for (int g = 0; g < 8; g++) {
      float f = rowScale[m0 + g];
      acc0.f[g] *= f;
      acc1.f[g] *= f;
    }
#pragma unroll
    for (int k = 0; k < 3; k++) {
      v16h a = afrag_rm(sP, 96, 0, k * 32, lane);
      v16h b0 = bfrag_nt(Vs, Tpad, 0,  s0 + k * 32, lane);
      v16h b1 = bfrag_nt(Vs, Tpad, 16, s0 + k * 32, lane);
      acc0.v = wmma32(a, b0, acc0.v);
      acc1.v = wmma32(a, b1, acc1.v);
    }
    __syncthreads();
  }

  float rTot = rSp + __shfl_xor(rSp, 16);          // full row sum
  if (lane < 16) rowSum[lane] = rTot;
  __syncthreads();

  long yb = ((long)seq * Tpad + q0 + m0) * HD + head * KD + (lane & 15);
#pragma unroll
  for (int g = 0; g < 8; g++) {
    float inv = 1.f / rowSum[m0 + g];
    Ym[yb + (long)g * HD]      = (_Float16)(acc0.f[g] * inv);
    Ym[yb + (long)g * HD + 16] = (_Float16)(acc1.f[g] * inv);
  }
}

// ---------------------------------------------------------------------------
// Output projection Y[.,128] @ Wp[128,96] + bp with fused epilogue.
// epi==0: out = val (intermediate axial pass)
// epi==1: out = base + coefConst*(*alphaPtr)*val
// ---------------------------------------------------------------------------
__global__ void proj_kernel(const _Float16* __restrict__ Ym,
                            const _Float16* __restrict__ WpP,
                            const float* __restrict__ bp,
                            const float* __restrict__ basep,
                            float* __restrict__ outp,
                            const float* __restrict__ alphaPtr, float coefConst,
                            int epi, int T, int Tpad, int Hd, int Wd,
                            int strided) {
  int lane = threadIdx.x;
  int strips = T >> 4;
  int seq = blockIdx.x / strips;
  int t0 = (blockIdx.x - seq * strips) << 4;
  float coef = coefConst * (alphaPtr ? alphaPtr[0] : 1.f);
  const _Float16* Yb = Ym + (long)seq * Tpad * HD;
  int m0 = (lane >> 4) << 3;

#pragma unroll
  for (int nt = 0; nt < 6; nt++) {
    V8F acc = {};
#pragma unroll
    for (int kt = 0; kt < 4; kt++) {
      v16h a = afrag_rm(Yb, HD, t0, kt * 32, lane);
      v16h b = bfrag_packed(WpP, kt * 6 + nt, lane);
      acc.v = wmma32(a, b, acc.v);
    }
    int n = nt * 16 + (lane & 15);
    float bb = bp[n];
#pragma unroll
    for (int g = 0; g < 8; g++) {
      int t = t0 + m0 + g;
      int idx = tok_base(strided, Hd, Wd, T, seq, t) + n;
      float val = acc.f[g] + bb;
      outp[idx] = epi ? (basep[idx] + coef * val) : val;
    }
  }
}

// ---------------------------------------------------------------------------
// Orientation mix (3x3) + strength-0.25 blend, in place over d_out slots.
// ---------------------------------------------------------------------------
__global__ void mix_blend_kernel(const float* __restrict__ h0,
                                 const float* __restrict__ h1,
                                 const float* __restrict__ h2,
                                 const float* __restrict__ mw,
                                 float* o0, float* o1, float* o2, int n) {
  int i = blockIdx.x * blockDim.x + threadIdx.x;
  if (i >= n) return;
  float p0 = o0[i], p1 = o1[i], p2 = o2[i];
  float y0 = mw[0] * p0 + mw[1] * p1 + mw[2] * p2;
  float y1 = mw[3] * p0 + mw[4] * p1 + mw[5] * p2;
  float y2 = mw[6] * p0 + mw[7] * p1 + mw[8] * p2;
  float a0 = h0[i], a1 = h1[i], a2 = h2[i];
  o0[i] = a0 + 0.25f * (y0 - a0);
  o1[i] = a1 + 0.25f * (y1 - a1);
  o2[i] = a2 + 0.25f * (y2 - a2);
}

// ---------------------------------------------------------------------------
extern "C" void kernel_launch(void* const* d_in, const int* in_sizes, int n_in,
                              void* d_out, int out_size, void* d_ws,
                              size_t ws_size, hipStream_t stream) {
  (void)in_sizes; (void)n_in; (void)out_size; (void)ws_size;
  const int B = 4;
  const float* in[42];
  for (int i = 0; i < 42; i++) in[i] = (const float*)d_in[i];
  float* out = (float*)d_out;

  const int S0 = B * 96 * 96 * CCH;  // 3538944
  const int S1 = B * 48 * 48 * CCH;  // 884736
  float* o_lp0 = out;
  float* o_hp0[3] = {out + (size_t)S0, out + 2 * (size_t)S0, out + 3 * (size_t)S0};
  float* o_lp1 = out + 4 * (size_t)S0;
  float* o_hp1[3] = {o_lp1 + (size_t)S1, o_lp1 + 2 * (size_t)S1,
                     o_lp1 + 3 * (size_t)S1};

  // workspace layout
  char* wsb = (char*)d_ws;
  _Float16* wPack = (_Float16*)wsb;                    // 3 sets * 96 frags * 512
  size_t off = 3ull * 96 * 512 * sizeof(_Float16);     // 294912 B
  const size_t QKVB = 36864ull * HD * sizeof(_Float16); // 9.44 MB each
  _Float16* Qb  = (_Float16*)(wsb + off); off += QKVB;
  _Float16* Kb  = (_Float16*)(wsb + off); off += QKVB;
  _Float16* Vtb = (_Float16*)(wsb + off); off += QKVB;
  _Float16* Yb  = (_Float16*)(wsb + off); off += QKVB;
  float* tmp = (float*)(wsb + off);                    // fp32 intermediate

  // --- pack weights (3 param sets: p_hp, p_lpa, p_lpg) ---------------------
  for (int s = 0; s < 3; s++) {
    int pb = 8 + 10 * s;
    _Float16* base = wPack + (size_t)s * 96 * 512;
    pack_w_kernel<<<24, 32, 0, stream>>>(in[pb + 2], base,            96, HD);
    pack_w_kernel<<<24, 32, 0, stream>>>(in[pb + 4], base + 24 * 512, 96, HD);
    pack_w_kernel<<<24, 32, 0, stream>>>(in[pb + 6], base + 48 * 512, 96, HD);
    pack_w_kernel<<<24, 32, 0, stream>>>(in[pb + 8], base + 72 * 512, HD, 96);
  }

  auto run_pass = [&](const float* xin, int s, int T, int Tpad, int nSeq,
                      int strided, int Hd, int Wd, float* outp,
                      const float* basep, const float* alphaP, float coefC,
                      int epi) {
    int pb = 8 + 10 * s;
    _Float16* wb = wPack + (size_t)s * 96 * 512;
    if (Tpad != T) {  // zero pad keys/values so padded WMMA lanes contribute 0
      (void)hipMemsetAsync(Kb, 0, (size_t)nSeq * Tpad * HD * 2, stream);
      (void)hipMemsetAsync(Vtb, 0, (size_t)nSeq * Tpad * HD * 2, stream);
    }
    ln_qkv_kernel<<<nSeq * (T / 16), 96, 0, stream>>>(
        xin, in[pb + 0], in[pb + 1], wb, in[pb + 3], wb + 24 * 512, in[pb + 5],
        wb + 48 * 512, in[pb + 7], Qb, Kb, Vtb, T, Tpad, Hd, Wd, strided);
    attn_kernel<<<nSeq * NHEAD * (T / 16), 32, 0, stream>>>(Qb, Kb, Vtb, Yb, T,
                                                            Tpad);
    proj_kernel<<<nSeq * (T / 16), 32, 0, stream>>>(
        Yb, wb + 72 * 512, in[pb + 9], basep, outp, alphaP, coefC, epi, T, Tpad,
        Hd, Wd, strided);
  };

  // axial = attend along W (seqs (b,h)) then along H (seqs (b,w)); the inner
  // pass output (pure projection, no residual) goes through `tmp`.
  auto axial = [&](const float* xin, int s, int Hd, int Wd, float* outp,
                   const float* basep, const float* alphaP, float coefC) {
    int T0 = Wd, T0p = (T0 < 96) ? 96 : T0;
    run_pass(xin, s, T0, T0p, B * Hd, 0, Hd, Wd, tmp, nullptr, nullptr, 0.f, 0);
    int T1 = Hd, T1p = (T1 < 96) ? 96 : T1;
    run_pass(tmp, s, T1, T1p, B * Wd, 1, Hd, Wd, outp, basep, alphaP, coefC, 1);
  };

  const float* alpha_hp = in[40];
  const float* alpha_lp = in[41];

  // ---- level 0 (scale = DECAY = 0.6), axial everywhere --------------------
  const float* hp0[3] = {in[1], in[2], in[3]};
  for (int i = 0; i < 3; i++)
    axial(hp0[i], 0, 96, 96, o_hp0[i], hp0[i], alpha_hp, 0.6f);  // pre-mix
  axial(in[0], 1, 96, 96, o_lp0, in[0], alpha_lp, 0.25f * 0.6f); // fused blend
  mix_blend_kernel<<<(S0 + 255) / 256, 256, 0, stream>>>(
      hp0[0], hp0[1], hp0[2], in[38], o_hp0[0], o_hp0[1], o_hp0[2], S0);

  // ---- level 1 (scale = 1.0), hp axial, lp global (T = 48*48 = 2304) ------
  const float* hp1[3] = {in[5], in[6], in[7]};
  for (int i = 0; i < 3; i++)
    axial(hp1[i], 0, 48, 48, o_hp1[i], hp1[i], alpha_hp, 1.0f);
  run_pass(in[4], 2, 2304, 2304, B, 0, 48, 48, o_lp1, in[4], alpha_lp, 0.25f, 1);
  mix_blend_kernel<<<(S1 + 255) / 256, 256, 0, stream>>>(
      hp1[0], hp1[1], hp1[2], in[39], o_hp1[0], o_hp1[1], o_hp1[2], S1);
}
